// Quantize_51634096832528
// MI455X (gfx1250) — compile-verified, matrix operand order checked by reading, not run on hardware
//
#include <hip/hip_runtime.h>
#include <cstdint>

// ---------------------------------------------------------------------------
// VQ nearest-codebook kernel for gfx1250 (MI455X).
// d = ||e||^2 - 2 x.e  (||x||^2 constant per row -> dropped for argmin)
// bf16x3 split-precision WMMA + double-buffered async global->LDS staging,
// software-pipelined argmin epilogue to hide WMMA->VALU hazards.
// ---------------------------------------------------------------------------

typedef __attribute__((ext_vector_type(16))) __bf16 v16bf;
typedef __attribute__((ext_vector_type(8)))  float  v8f;

#define DIM      64     // embedding dim
#define NEMB     1024   // codebook size
#define KTILE    128    // embeddings staged in LDS per iteration
#define NTILES   (KTILE / 16)
#define NKT      (NEMB / KTILE)
#define WAVES    8
#define BLOCK    (WAVES * 32)

static __device__ inline v8f wmma_bf16(v16bf a, v16bf b, v8f c) {
    // (neg_a, A, neg_b, B, c_mod, C, reuse_a, reuse_b)
    return __builtin_amdgcn_wmma_f32_16x16x32_bf16(
        false, a, false, b, (short)0, c, false, false);
}

// Low 32 bits of a generic pointer into LDS == wave-relative LDS address.
static __device__ inline uint32_t lds_addr(const void* p) {
    return (uint32_t)(uintptr_t)p;
}

// One 16-byte async global->LDS copy per lane (ASYNCcnt-tracked).
static __device__ inline void async_b128(uint32_t lds_off, uint32_t voff,
                                         const void* sbase) {
    asm volatile("global_load_async_to_lds_b128 %0, %1, %2"
                 :: "v"(lds_off), "v"(voff),
                    "s"((unsigned long long)(uintptr_t)sbase)
                 : "memory");
}

static __device__ inline void wait_async0() {
    asm volatile("s_wait_asynccnt 0x0" ::: "memory");
}

// ---------------------------------------------------------------------------
// Kernel 1: one-time codebook prep: ||e||^2 and bf16 hi/lo split codebooks.
// ---------------------------------------------------------------------------
__global__ void vq_prep_kernel(const float* __restrict__ emb,
                               float* __restrict__ enorm,
                               __bf16* __restrict__ embh,
                               __bf16* __restrict__ embl) {
    int k = blockIdx.x * blockDim.x + threadIdx.x;
    if (k >= NEMB) return;
    const float4* p = (const float4*)(emb + (size_t)k * DIM);
    float s = 0.0f;
#pragma unroll
    for (int i = 0; i < DIM / 4; ++i) {
        float4 v = p[i];
        float t[4] = {v.x, v.y, v.z, v.w};
#pragma unroll
        for (int j = 0; j < 4; ++j) {
            float  f = t[j];
            __bf16 h = (__bf16)f;
            embh[(size_t)k * DIM + i * 4 + j] = h;
            embl[(size_t)k * DIM + i * 4 + j] = (__bf16)(f - (float)h);
            s += f * f;
        }
    }
    enorm[k] = s;
}

// ---------------------------------------------------------------------------
// Kernel 2: argmin over codebook via bf16x3 WMMA, double-buffered async LDS.
// One wave owns a 16-row tile of x (kept in registers as WMMA A fragments).
// ---------------------------------------------------------------------------
__global__ __launch_bounds__(BLOCK) void vq_argmin_kernel(
    const float*  __restrict__ x,
    const float*  __restrict__ emb,
    const float*  __restrict__ enorm,
    const __bf16* __restrict__ embh,
    const __bf16* __restrict__ embl,
    float* __restrict__ qout,      // [N, 64] quantized output
    float* __restrict__ idxout,    // [N] argmin indices (as float)
    int nrows)
{
    __shared__ __attribute__((aligned(16))) __bf16 s_hi[2][KTILE * DIM];
    __shared__ __attribute__((aligned(16))) __bf16 s_lo[2][KTILE * DIM];
    __shared__ float s_norm[2][KTILE];

    const int tid  = threadIdx.x;
    const int lane = tid & 31;
    const int wave = tid >> 5;
    const int row0 = (blockIdx.x * WAVES + wave) * 16;
    if (row0 >= nrows) return;   // never taken for full grid; keeps code safe

    // ---- A fragments: 16 rows x 64 K, bf16 hi/lo, two K-chunks of 32 ------
    // ISA 16-bit A 16x32 layout: lane m=lane&15; lanes<16 hold K 0-7,16-23;
    // lanes>=16 hold K 8-15,24-31  => elem e: K = kb + e + (e>=8)*8
    const int m  = lane & 15;
    const int kb = (lane >> 4) << 3;

    v16bf a_hi[2], a_lo[2];
    {
        const float* xrow = x + (size_t)(row0 + m) * DIM;
#pragma unroll
        for (int c = 0; c < 2; ++c) {
            const float4* p0 = (const float4*)(xrow + c * 32 + kb);
            const float4* p1 = (const float4*)(xrow + c * 32 + kb + 16);
            float4 v0 = p0[0], v1 = p0[1], v2 = p1[0], v3 = p1[1];
            float t[16];
            t[0]=v0.x; t[1]=v0.y; t[2]=v0.z; t[3]=v0.w;
            t[4]=v1.x; t[5]=v1.y; t[6]=v1.z; t[7]=v1.w;
            t[8]=v2.x; t[9]=v2.y; t[10]=v2.z; t[11]=v2.w;
            t[12]=v3.x; t[13]=v3.y; t[14]=v3.z; t[15]=v3.w;
#pragma unroll
            for (int e = 0; e < 16; ++e) {
                __bf16 h = (__bf16)t[e];
                a_hi[c][e] = h;
                a_lo[c][e] = (__bf16)(t[e] - (float)h);
            }
        }
    }

    float minval[8];
    int   minidx[8];
#pragma unroll
    for (int r = 0; r < 8; ++r) { minval[r] = 3.4e38f; minidx[r] = 0; }

    // ---- prologue: async-stage tile 0 into buffer 0 ------------------------
    // Each hi/lo buffer is KTILE*DIM*2 = 16 KB -> 4 b128 per thread per array.
#define STAGE_ITERS ((KTILE * DIM * 2) / (BLOCK * 16))
    {
        const uint32_t h0 = lds_addr(&s_hi[0][0]);
        const uint32_t l0 = lds_addr(&s_lo[0][0]);
#pragma unroll
        for (int it = 0; it < STAGE_ITERS; ++it) {
            uint32_t boff = (uint32_t)(tid + it * BLOCK) * 16u;
            async_b128(h0 + boff, boff, embh);
            async_b128(l0 + boff, boff, embl);
        }
        if (tid < KTILE) s_norm[0][tid] = enorm[tid];
    }

    for (int kt = 0; kt < NKT; ++kt) {
        const int b = kt & 1;
        wait_async0();        // my async writes into buf b have landed
        __syncthreads();      // everyone's landed; prev readers of buf b^1 done

        if (kt + 1 < NKT) {   // stream next tile into the other buffer
            const int nb = b ^ 1;
            const __bf16* gh = embh + (size_t)(kt + 1) * KTILE * DIM;
            const __bf16* gl = embl + (size_t)(kt + 1) * KTILE * DIM;
            const uint32_t hb = lds_addr(&s_hi[nb][0]);
            const uint32_t lb = lds_addr(&s_lo[nb][0]);
#pragma unroll
            for (int it = 0; it < STAGE_ITERS; ++it) {
                uint32_t boff = (uint32_t)(tid + it * BLOCK) * 16u;
                async_b128(hb + boff, boff, gh);
                async_b128(lb + boff, boff, gl);
            }
            if (tid < KTILE) s_norm[nb][tid] = enorm[(kt + 1) * KTILE + tid];
        }

        // ---- compute on buffer b, epilogue pipelined one tile behind ------
        // Tile nt's 6 WMMAs issue first; tile nt-1's distance/argmin VALU
        // then co-executes with them (its WMMA->VALU hazard long expired).
        v8f   pacc0 = {}, pacc1 = {};
        float pen   = 0.0f;
        int   peidx = 0;
#pragma unroll
        for (int nt = 0; nt < NTILES + 1; ++nt) {
            v8f   acc0 = {}, acc1 = {};
            float en   = 0.0f;
            int   eidx = 0;

            if (nt < NTILES) {
                // B fragments from LDS (emb row-major => B = emb^T)
                v16bf b_hi[2], b_lo[2];
#pragma unroll
                for (int c = 0; c < 2; ++c) {
                    const __bf16* ph = &s_hi[b][(nt*16 + m)*DIM + c*32 + kb];
                    const __bf16* pl = &s_lo[b][(nt*16 + m)*DIM + c*32 + kb];
#pragma unroll
                    for (int e = 0; e < 8; ++e) {
                        b_hi[c][e]     = ph[e];
                        b_hi[c][8 + e] = ph[16 + e];
                        b_lo[c][e]     = pl[e];
                        b_lo[c][8 + e] = pl[16 + e];
                    }
                }
                en   = s_norm[b][nt * 16 + m];
                eidx = kt * KTILE + nt * 16 + m;

                // bf16x3: hh -> acc0, hl+lh -> acc1 (two independent chains)
#pragma unroll
                for (int c = 0; c < 2; ++c) {
                    acc0 = wmma_bf16(a_hi[c], b_hi[c], acc0);
                    acc1 = wmma_bf16(a_hi[c], b_lo[c], acc1);
                    acc1 = wmma_bf16(a_lo[c], b_hi[c], acc1);
                }
            }

            if (nt > 0) {
                // distance + running argmin for the PREVIOUS tile
#pragma unroll
                for (int r = 0; r < 8; ++r) {
                    float dv = pen - 2.0f * (pacc0[r] + pacc1[r]);
                    if (dv < minval[r]) { minval[r] = dv; minidx[r] = peidx; }
                }
            }

            pacc0 = acc0;
            pacc1 = acc1;
            pen   = en;
            peidx = eidx;
        }
    }

    // ---- argmin reduction across the 16 lanes holding one row -------------
    // C/D layout: lane group g=lane>>4 holds rows M = r + 8*g, cols = lane&15.
#pragma unroll
    for (int r = 0; r < 8; ++r) {
        float v  = minval[r];
        int   id = minidx[r];
#pragma unroll
        for (int off = 8; off >= 1; off >>= 1) {
            float ov  = __shfl_xor(v,  off, 16);
            int   oid = __shfl_xor(id, off, 16);
            if (ov < v || (ov == v && oid < id)) { v = ov; id = oid; }
        }
        minval[r] = v;
        minidx[r] = id;
    }

    // Broadcast all 16 row winners to every lane of the wave.
    int fidx[16];
#pragma unroll
    for (int r = 0; r < 8; ++r) {
        fidx[r]     = __shfl(minidx[r], 0,  32);  // rows 0..7   (group 0)
        fidx[8 + r] = __shfl(minidx[r], 16, 32);  // rows 8..15  (group 1)
    }

    // ---- write indices (coalesced) ----------------------------------------
    if (lane < 16)
        idxout[row0 + lane] = (float)fidx[lane];

    // ---- gather winning embedding rows, 2 floats/lane, coalesced ----------
    const float2* embv = (const float2*)emb;
    float2*       qv   = (float2*)qout;
#pragma unroll
    for (int mm = 0; mm < 16; ++mm) {
        int e = fidx[mm];
        qv[(size_t)(row0 + mm) * (DIM / 2) + lane] =
            embv[(size_t)e * (DIM / 2) + lane];
    }
}

// ---------------------------------------------------------------------------
// Launch
// ---------------------------------------------------------------------------
extern "C" void kernel_launch(void* const* d_in, const int* in_sizes, int n_in,
                              void* d_out, int out_size, void* d_ws, size_t ws_size,
                              hipStream_t stream) {
    const float* x   = (const float*)d_in[0];   // [32, 64, 4096] f32
    const float* emb = (const float*)d_in[1];   // [1024, 64] f32
    float* out = (float*)d_out;

    // workspace layout: enorm f32[1024] | embh bf16[1024*64] | embl bf16[1024*64]
    char* ws = (char*)d_ws;
    float*  enorm = (float*)ws;                       //   4 KB
    __bf16* embh  = (__bf16*)(ws + 4096);             // 128 KB
    __bf16* embl  = (__bf16*)(ws + 4096 + (size_t)NEMB * DIM * 2);

    const int nrows = in_sizes[0] / DIM;              // 131072
    float* qout   = out;                              // [nrows * 64]
    float* idxout = out + (size_t)nrows * DIM;        // [nrows]

    vq_prep_kernel<<<(NEMB + 255) / 256, 256, 0, stream>>>(emb, enorm, embh, embl);

    const int blocks = (nrows / 16) / WAVES;          // 1024
    vq_argmin_kernel<<<blocks, BLOCK, 0, stream>>>(x, emb, enorm, embh, embl,
                                                   qout, idxout, nrows);
}
